// LSTM_10445360464276
// MI455X (gfx1250) — matrix operand order repaired and checked by reading, not compiled, not run on hardware
//
#include <hip/hip_runtime.h>
#include <hip/hip_bf16.h>

typedef __attribute__((ext_vector_type(16))) _Float16 v16h;
typedef __attribute__((ext_vector_type(8)))  _Float16 v8h;
typedef __attribute__((ext_vector_type(8)))  float    v8f;

#define B_   256
#define F_   8
#define T_   512
#define H_   512
#define G_   2048          // 4*H
#define NBLK 128
#define NTHR 128           // 4 waves of 32
#define SLOT (B_*H_)       // halves per ping-pong slot
#define LDSH 520           // padded LDS row stride (halves): 1040B = 260 dwords, 260%64=4 -> conflict-free

__device__ __forceinline__ float rcp_(float x){ return __builtin_amdgcn_rcpf(x); }
__device__ __forceinline__ float sigmoidf_(float x){ return rcp_(1.f + __expf(-x)); }
__device__ __forceinline__ float tanhf_(float x){ return 1.f - 2.f*rcp_(__expf(2.f*x) + 1.f); }

// B-fragment: 32(K) x 16(N). lane -> col = nbase+(lane&15); K span = (lane>=16 ? 16..31 : 0..15)
__device__ __forceinline__ v16h load_bfrag(const _Float16* __restrict__ W, int ldk,
                                           int nbase, int kc, int lane){
  int n = nbase + (lane & 15);
  const _Float16* p = W + (size_t)n*ldk + kc + ((lane >> 4) << 4);
  v8h lo = *(const v8h*)p;
  v8h hi = *(const v8h*)(p + 8);
  v16h r;
  #pragma unroll
  for (int i = 0; i < 8; ++i){ r[i] = lo[i]; r[i+8] = hi[i]; }
  return r;
}

// A-fragment from LDS tile (16 local rows, stride LDSH halves).
// lanes 0-15: K = kc+{0..15}; lanes 16-31: K = kc+{16..31}
__device__ __forceinline__ v16h load_afrag_lds(const _Float16* sh, int kc, int lane){
  int row = lane & 15;
  int kb  = (lane >> 4) << 4;
  const _Float16* p = sh + row*LDSH + kc + kb;
  v8h lo = *(const v8h*)p;
  v8h hi = *(const v8h*)(p + 8);
  v16h r;
  #pragma unroll
  for (int i = 0; i < 8; ++i){ r[i] = lo[i]; r[i+8] = hi[i]; }
  return r;
}

// Async-copy one 16-row x 512-half tile (16KB, rows contiguous in global) into a
// padded LDS tile. 128 threads x 8 instructions x 16B. The 24-bit IOFFSET is added
// to BOTH the LDS and global addresses (ISA 15.18.3), so one base pair serves all 8.
__device__ __forceinline__ void async_tile_copy(_Float16* shdst, const _Float16* gsrc, int tid){
  int row = tid >> 3;
  int sub = (tid & 7) << 4;                                   // 0..112 bytes within row
  unsigned loff = (unsigned)(size_t)(shdst) + (unsigned)(row*(LDSH*2) + sub);
  const char* g = (const char*)(gsrc + (size_t)row*H_) + sub; // global row stride 1024B
  asm volatile(
    "global_load_async_to_lds_b128 %0, %1, off\n\t"
    "global_load_async_to_lds_b128 %0, %1, off offset:128\n\t"
    "global_load_async_to_lds_b128 %0, %1, off offset:256\n\t"
    "global_load_async_to_lds_b128 %0, %1, off offset:384\n\t"
    "global_load_async_to_lds_b128 %0, %1, off offset:512\n\t"
    "global_load_async_to_lds_b128 %0, %1, off offset:640\n\t"
    "global_load_async_to_lds_b128 %0, %1, off offset:768\n\t"
    "global_load_async_to_lds_b128 %0, %1, off offset:896"
    :: "v"(loff), "v"(g)
    : "memory");
}

__device__ __forceinline__ void grid_sync(unsigned* bar, unsigned target){
  __syncthreads();
  if (threadIdx.x == 0){
    __threadfence();
    unsigned prev = __hip_atomic_fetch_add(&bar[0], 1u, __ATOMIC_ACQ_REL, __HIP_MEMORY_SCOPE_AGENT);
    if (prev == (unsigned)(NBLK - 1)){
      __hip_atomic_store(&bar[0], 0u, __ATOMIC_RELAXED, __HIP_MEMORY_SCOPE_AGENT);
      __hip_atomic_store(&bar[1], target, __ATOMIC_RELEASE, __HIP_MEMORY_SCOPE_AGENT);
    } else {
      while (__hip_atomic_load(&bar[1], __ATOMIC_ACQUIRE, __HIP_MEMORY_SCOPE_AGENT) < target){
        __builtin_amdgcn_s_sleep(1);
      }
    }
  }
  __syncthreads();
}

#define WMMA_F16(A, Bm, C) \
  __builtin_amdgcn_wmma_f32_16x16x32_f16(false, (A), false, (Bm), (short)0, (C), false, false)

// ---------------- init: convert weights to f16, fuse biases, zero state ----------------
__global__ void lstm_init_kernel(const float* __restrict__ Wih1, const float* __restrict__ Whh1,
                                 const float* __restrict__ Wih2, const float* __restrict__ Whh2,
                                 const float* __restrict__ bih1, const float* __restrict__ bhh1,
                                 const float* __restrict__ bih2, const float* __restrict__ bhh2,
                                 _Float16* __restrict__ Wih1p, _Float16* __restrict__ Whh1h,
                                 _Float16* __restrict__ Wih2h, _Float16* __restrict__ Whh2h,
                                 float* __restrict__ bias1, float* __restrict__ bias2,
                                 _Float16* __restrict__ h1buf, _Float16* __restrict__ h2buf,
                                 unsigned* __restrict__ bar){
  size_t i0 = (size_t)blockIdx.x*blockDim.x + threadIdx.x;
  size_t st = (size_t)gridDim.x*blockDim.x;
  if (i0 == 0){ bar[0] = 0u; bar[1] = 0u; }
  for (size_t i = i0; i < (size_t)G_*32; i += st){
    size_t n = i >> 5, k = i & 31;
    Wih1p[i] = (k < F_) ? (_Float16)Wih1[n*F_ + k] : (_Float16)0.f;
  }
  for (size_t i = i0; i < (size_t)G_*H_; i += st) Whh1h[i] = (_Float16)Whh1[i];
  for (size_t i = i0; i < (size_t)G_*H_; i += st) Wih2h[i] = (_Float16)Wih2[i];
  for (size_t i = i0; i < (size_t)G_*H_; i += st) Whh2h[i] = (_Float16)Whh2[i];
  for (size_t i = i0; i < (size_t)G_;    i += st){ bias1[i] = bih1[i]+bhh1[i]; bias2[i] = bih2[i]+bhh2[i]; }
  for (size_t i = i0; i < 2*(size_t)SLOT; i += st){ h1buf[i] = (_Float16)0.f; h2buf[i] = (_Float16)0.f; }
}

// ---------------- persistent 2-layer LSTM over all T timesteps ----------------
__launch_bounds__(NTHR, 1)
__global__ void lstm_persistent_kernel(
    const float* __restrict__ y,           // (B, F, T)
    const _Float16* __restrict__ Wih1p,    // (G, 32) zero-padded
    const _Float16* __restrict__ Whh1,     // (G, H)
    const _Float16* __restrict__ Wih2,     // (G, H)
    const _Float16* __restrict__ Whh2,     // (G, H)
    const float* __restrict__ bias1,       // (G)
    const float* __restrict__ bias2,       // (G)
    const float* __restrict__ wlin,        // (H)
    const float* __restrict__ blin,        // (1)
    _Float16* __restrict__ h1buf,          // 2*SLOT ping-pong
    _Float16* __restrict__ h2buf,          // 2*SLOT ping-pong
    unsigned* __restrict__ bar,
    float* __restrict__ out)               // (B, T)
{
  __shared__ _Float16 shA[16*LDSH];        // h1(t-1) tile for this block's mtile
  __shared__ _Float16 shB[16*LDSH];        // h1(t)   tile
  __shared__ _Float16 shC[16*LDSH];        // h2(t-1) tile

  const int tid   = threadIdx.x;
  const int lane  = tid & 31;
  const int wave  = tid >> 5;
  const int mtile = blockIdx.x >> 3;                 // 0..15 (shared by the 4 waves)
  const int jtile = ((blockIdx.x & 7) << 2) | wave;  // 0..31
  const int mbase = mtile << 4;
  const int jbase = jtile << 4;
  const int col   = jbase + (lane & 15);
  const int rofs  = (lane >> 4) << 3;                // +8 rows for hi half-wave

  v8f c1; v8f c2;
  #pragma unroll
  for (int e = 0; e < 8; ++e){ c1[e] = 0.f; c2[e] = 0.f; }

  float b1g[4], b2g[4];
  #pragma unroll
  for (int g = 0; g < 4; ++g){ b1g[g] = bias1[g*H_ + col]; b2g[g] = bias2[g*H_ + col]; }
  const float bl = blin[0];

  unsigned target = 1;

  #pragma unroll 1
  for (int t = 0; t < T_; ++t){
    const _Float16* h1r = h1buf + (size_t)((t & 1) ^ 1) * SLOT;
    _Float16*       h1w = h1buf + (size_t)((t & 1)    ) * SLOT;
    const _Float16* h2r = h2buf + (size_t)((t & 1) ^ 1) * SLOT;
    _Float16*       h2w = h2buf + (size_t)((t & 1)    ) * SLOT;

    // Stage h1(t-1) and h2(t-1) row-tiles into LDS asynchronously; h2 copy
    // overlaps all of layer-1 compute (ASYNCcnt completes in order).
    async_tile_copy(shA, h1r + (size_t)mbase*H_, tid);
    async_tile_copy(shC, h2r + (size_t)mbase*H_, tid);
    asm volatile("s_wait_asynccnt 0x8" ::: "memory");   // shA (this wave) done
    __syncthreads();                                    // shA done block-wide

    // =========== layer 1: gates = x@Wih1^T + h1@Whh1^T + b ===========
    v8f a0, a1, a2, a3;
    #pragma unroll
    for (int e = 0; e < 8; ++e){ a0[e] = b1g[0]; a1[e] = b1g[1]; a2[e] = b1g[2]; a3[e] = b1g[3]; }

    // x chunk: A rows carry x_t in K 0..7, zeros elsewhere (padded Wih1 matches)
    v16h ax;
    #pragma unroll
    for (int i = 0; i < 16; ++i) ax[i] = (_Float16)0.f;
    if (lane < 16){
      int row = mbase + lane;
      const float* xp = y + (size_t)row*F_*T_ + t;
      #pragma unroll
      for (int f = 0; f < F_; ++f) ax[f] = (_Float16)xp[(size_t)f*T_];
    }
    {
      v16h b0 = load_bfrag(Wih1p, 32, 0*H_ + jbase, 0, lane);
      v16h b1 = load_bfrag(Wih1p, 32, 1*H_ + jbase, 0, lane);
      v16h b2 = load_bfrag(Wih1p, 32, 2*H_ + jbase, 0, lane);
      v16h b3 = load_bfrag(Wih1p, 32, 3*H_ + jbase, 0, lane);
      a0 = WMMA_F16(ax, b0, a0);
      a1 = WMMA_F16(ax, b1, a1);
      a2 = WMMA_F16(ax, b2, a2);
      a3 = WMMA_F16(ax, b3, a3);
    }
    #pragma unroll 4
    for (int c = 0; c < 16; ++c){
      int kc = c << 5;
      __builtin_prefetch(Whh1 + (size_t)(jbase + (lane & 15))*H_ + ((kc + 32) & (H_ - 1)), 0, 3);
      v16h a  = load_afrag_lds(shA, kc, lane);
      v16h b0 = load_bfrag(Whh1, H_, 0*H_ + jbase, kc, lane);
      v16h b1 = load_bfrag(Whh1, H_, 1*H_ + jbase, kc, lane);
      v16h b2 = load_bfrag(Whh1, H_, 2*H_ + jbase, kc, lane);
      v16h b3 = load_bfrag(Whh1, H_, 3*H_ + jbase, kc, lane);
      a0 = WMMA_F16(a, b0, a0);
      a1 = WMMA_F16(a, b1, a1);
      a2 = WMMA_F16(a, b2, a2);
      a3 = WMMA_F16(a, b3, a3);
    }
    #pragma unroll
    for (int e = 0; e < 8; ++e){
      float ig = sigmoidf_(a0[e]);
      float fg = sigmoidf_(a1[e]);
      float gg = tanhf_(a2[e]);
      float og = sigmoidf_(a3[e]);
      float cn = fg*c1[e] + ig*gg;
      c1[e] = cn;
      float hv = og * tanhf_(cn);
      int row = mbase + e + rofs;
      h1w[(size_t)row*H_ + col] = (_Float16)hv;
    }
    grid_sync(bar, target++);   // all of h1(t) visible device-wide

    // Stage h1(t) tile; wait for it and the earlier h2(t-1) copy.
    async_tile_copy(shB, h1w + (size_t)mbase*H_, tid);
    asm volatile("s_wait_asynccnt 0x0" ::: "memory");
    __syncthreads();

    // =========== layer 2: gates = h1@Wih2^T + h2@Whh2^T + b ===========
    #pragma unroll
    for (int e = 0; e < 8; ++e){ a0[e] = b2g[0]; a1[e] = b2g[1]; a2[e] = b2g[2]; a3[e] = b2g[3]; }
    #pragma unroll 4
    for (int c = 0; c < 16; ++c){
      int kc = c << 5;
      v16h a  = load_afrag_lds(shB, kc, lane);
      v16h b0 = load_bfrag(Wih2, H_, 0*H_ + jbase, kc, lane);
      v16h b1 = load_bfrag(Wih2, H_, 1*H_ + jbase, kc, lane);
      v16h b2 = load_bfrag(Wih2, H_, 2*H_ + jbase, kc, lane);
      v16h b3 = load_bfrag(Wih2, H_, 3*H_ + jbase, kc, lane);
      a0 = WMMA_F16(a, b0, a0);
      a1 = WMMA_F16(a, b1, a1);
      a2 = WMMA_F16(a, b2, a2);
      a3 = WMMA_F16(a, b3, a3);
    }
    #pragma unroll 4
    for (int c = 0; c < 16; ++c){
      int kc = c << 5;
      v16h a  = load_afrag_lds(shC, kc, lane);
      v16h b0 = load_bfrag(Whh2, H_, 0*H_ + jbase, kc, lane);
      v16h b1 = load_bfrag(Whh2, H_, 1*H_ + jbase, kc, lane);
      v16h b2 = load_bfrag(Whh2, H_, 2*H_ + jbase, kc, lane);
      v16h b3 = load_bfrag(Whh2, H_, 3*H_ + jbase, kc, lane);
      a0 = WMMA_F16(a, b0, a0);
      a1 = WMMA_F16(a, b1, a1);
      a2 = WMMA_F16(a, b2, a2);
      a3 = WMMA_F16(a, b3, a3);
    }
    #pragma unroll
    for (int e = 0; e < 8; ++e){
      float ig = sigmoidf_(a0[e]);
      float fg = sigmoidf_(a1[e]);
      float gg = tanhf_(a2[e]);
      float og = sigmoidf_(a3[e]);
      float cn = fg*c2[e] + ig*gg;
      c2[e] = cn;
      float hv = og * tanhf_(cn);
      int row = mbase + e + rofs;
      h2w[(size_t)row*H_ + col] = (_Float16)hv;
    }
    grid_sync(bar, target++);   // all of h2(t) visible device-wide

    // =========== projection: out[:, t] = h2 @ wlin + blin (16 waves) ===========
    if (jtile == 0){
      int row  = mbase + (lane & 15);
      int hsel = lane >> 4;
      const _Float16* hp = h2w + (size_t)row*H_ + hsel*256;
      const float*    wp = wlin + hsel*256;
      float s = 0.f;
      #pragma unroll 8
      for (int k = 0; k < 256; ++k) s += (float)hp[k] * wp[k];
      s += __shfl_xor(s, 16, 32);
      if (lane < 16) out[(size_t)row*T_ + t] = s + bl;
    }
  }
}

extern "C" void kernel_launch(void* const* d_in, const int* in_sizes, int n_in,
                              void* d_out, int out_size, void* d_ws, size_t ws_size,
                              hipStream_t stream) {
  (void)in_sizes; (void)n_in; (void)out_size; (void)ws_size;
  const float* y    = (const float*)d_in[0];
  // d_in[1] = secondary_variables (unused on this path)
  const float* Wih1 = (const float*)d_in[2];
  const float* Whh1 = (const float*)d_in[3];
  const float* bih1 = (const float*)d_in[4];
  const float* bhh1 = (const float*)d_in[5];
  const float* Wih2 = (const float*)d_in[6];
  const float* Whh2 = (const float*)d_in[7];
  const float* bih2 = (const float*)d_in[8];
  const float* bhh2 = (const float*)d_in[9];
  const float* wlin = (const float*)d_in[10];
  const float* blin = (const float*)d_in[11];
  float* out = (float*)d_out;

  char* ws = (char*)d_ws;
  size_t off = 0;
  auto alloc = [&](size_t bytes) -> void* {
    void* p = ws + off;
    off = (off + bytes + 255) & ~(size_t)255;
    return p;
  };
  unsigned*  bar    = (unsigned*)alloc(256);
  _Float16*  Wih1p  = (_Float16*)alloc((size_t)G_*32*sizeof(_Float16));
  _Float16*  Whh1h  = (_Float16*)alloc((size_t)G_*H_*sizeof(_Float16));
  _Float16*  Wih2h  = (_Float16*)alloc((size_t)G_*H_*sizeof(_Float16));
  _Float16*  Whh2h  = (_Float16*)alloc((size_t)G_*H_*sizeof(_Float16));
  float*     bias1  = (float*)alloc((size_t)G_*sizeof(float));
  float*     bias2  = (float*)alloc((size_t)G_*sizeof(float));
  _Float16*  h1buf  = (_Float16*)alloc((size_t)2*SLOT*sizeof(_Float16));
  _Float16*  h2buf  = (_Float16*)alloc((size_t)2*SLOT*sizeof(_Float16));

  lstm_init_kernel<<<1024, 256, 0, stream>>>(Wih1, Whh1, Wih2, Whh2,
                                             bih1, bhh1, bih2, bhh2,
                                             Wih1p, Whh1h, Wih2h, Whh2h,
                                             bias1, bias2, h1buf, h2buf, bar);

  lstm_persistent_kernel<<<NBLK, NTHR, 0, stream>>>(y, Wih1p, Whh1h, Wih2h, Whh2h,
                                                    bias1, bias2, wlin, blin,
                                                    h1buf, h2buf, bar, out);
}
